// MeshDownConv_49383533969436
// MI455X (gfx1250) — matrix operand order repaired and checked by reading, not compiled, not run on hardware
//
#include <hip/hip_runtime.h>
#include <hip/hip_bf16.h>

// ---------------------------------------------------------------------------
// MeshCNN down-conv stack for MI455X (gfx1250, wave32, WMMA).
// conv -> instance-norm -> relu (+residual), 3 convs total.
// GEMM per conv: M = edges, N = 64, K = C*5 (160 / 320), via
// v_wmma_f32_16x16x32_f16 with f32 accumulation. Weight panel staged into
// LDS with global_load_async_to_lds_b128 (ASYNCcnt) when available.
// ---------------------------------------------------------------------------

typedef __attribute__((ext_vector_type(16))) _Float16 v16h;
typedef __attribute__((ext_vector_type(2)))  _Float16 v2h;
typedef __attribute__((ext_vector_type(8)))  float    v8f;

#define B_    4
#define E_    50000
#define COUT  64
#define NW    4       // waves per conv block
#define EPS_  1e-5f

#if __has_builtin(__builtin_amdgcn_global_load_async_to_lds_b128) && \
    __has_builtin(__builtin_amdgcn_s_wait_asynccnt)
#define USE_ASYNC_LDS 1
typedef __attribute__((__vector_size__(16))) int v4i_;
typedef v4i_ __attribute__((address_space(1)))* as1_v4i;
typedef v4i_ __attribute__((address_space(3)))* as3_v4i;
#else
#define USE_ASYNC_LDS 0
#endif

union V16HU { v16h v; uint4 u[2]; };

// ---------------------------------------------------------------------------
// Repack W [O][C][5] f32 -> Wp [O][K] f16 with k = s*C + c  (K = C*5).
// ---------------------------------------------------------------------------
__global__ void pack_w_kernel(const float* __restrict__ W, _Float16* __restrict__ Wp,
                              int O, int C) {
    int n = O * C * 5;
    int i = blockIdx.x * blockDim.x + threadIdx.x;
    if (i >= n) return;
    int o = i / (C * 5);
    int r = i - o * (C * 5);
    int c = r / 5;
    int s = r - c * 5;
    Wp[o * (C * 5) + s * C + c] = (_Float16)W[i];
}

// ---------------------------------------------------------------------------
// Mesh conv via WMMA. Block = 128 threads (4 waves); each wave computes a
// 16-edge x 64-output tile. K = C*5 is a multiple of 32.
// LDS: [64][K] f16 weights, [NW][16][K] f16 feature tiles, [NW][16][4] i32 nbrs.
// ---------------------------------------------------------------------------
template <int C>
__global__ __launch_bounds__(NW * 32)
void mesh_conv_wmma(const float* __restrict__ x,      // [B][C][E]
                    const int*   __restrict__ eidx,   // [B][E][4]
                    const _Float16* __restrict__ Wp,  // [64][K] packed
                    const float* __restrict__ bias,   // [64]
                    float* __restrict__ y)            // [B][64][E]
{
    constexpr int K = C * 5;
    const int b    = blockIdx.y;
    const int wave = threadIdx.x >> 5;
    const int lane = threadIdx.x & 31;

    extern __shared__ char smem[];
    _Float16* w_lds = (_Float16*)smem;                 // 64*K halves
    _Float16* a_lds = w_lds + 64 * K;                  // NW*16*K halves
    int*      nbr   = (int*)(a_lds + NW * 16 * K);     // NW*16*4 ints

    // Stage packed weights (layout already matches; pure 16B copies).
    {
        const int n16 = (64 * K) / 8;                  // 8 halves per 16B
#if USE_ASYNC_LDS
        const uintptr_t g = (uintptr_t)Wp;
        const uint32_t  l = (uint32_t)(uintptr_t)w_lds;
        for (int i = threadIdx.x; i < n16; i += blockDim.x) {
            __builtin_amdgcn_global_load_async_to_lds_b128(
                (as1_v4i)(g + (size_t)i * 16),
                (as3_v4i)(uintptr_t)(l + (uint32_t)i * 16),
                0, 0);
        }
#else
        const uint4* src = (const uint4*)Wp;
        uint4*       dst = (uint4*)w_lds;
        for (int i = threadIdx.x; i < n16; i += blockDim.x) dst[i] = src[i];
#endif
    }

    const int tiles = E_ / 16;                         // 3125 (exact)
    const int t     = blockIdx.x * NW + wave;
    const bool valid = (t < tiles);                    // wave-uniform -> EXEC all-1s
    const int e0    = valid ? t * 16 : 0;

    // Stage neighbor indices: one int4 per edge.
    if (valid && lane < 16) {
        const int4 v = ((const int4*)eidx)[(size_t)b * E_ + e0 + lane];
        int* p = nbr + (wave * 16 + lane) * 4;
        p[0] = v.x; p[1] = v.y; p[2] = v.z; p[3] = v.w;
    }
    __syncthreads();   // nbr visible (w_lds copy may still be in flight if async)

    // Build the 16 x K feature tile: feats = [x, n1+n3, n2+n4, |n1-n3|, |n2-n4|],
    // stored at k = s*C + c (matches packed W). Two consecutive channels per
    // step -> 32-bit LDS stores.
    const float* xb = x + (size_t)b * C * E_;
    if (valid) {
        constexpr int C2 = C / 2;
        _Float16* arow = a_lds + wave * 16 * K;
        for (int p = lane; p < 16 * C2; p += 32) {
            const int m  = p / C2;
            const int c  = (p - m * C2) * 2;
            const int* nb = nbr + (wave * 16 + m) * 4;
            const float* xc0 = xb + (size_t)c * E_;
            const float* xc1 = xc0 + E_;
            const float a0 = xc0[e0 + m],  a1 = xc1[e0 + m];
            const float p0 = xc0[nb[0]],   p1 = xc1[nb[0]];
            const float q0 = xc0[nb[1]],   q1 = xc1[nb[1]];
            const float r0 = xc0[nb[2]],   r1 = xc1[nb[2]];
            const float s0 = xc0[nb[3]],   s1 = xc1[nb[3]];
            _Float16* d = arow + m * K + c;
            *(v2h*)(d + 0 * C) = (v2h){(_Float16)a0,              (_Float16)a1};
            *(v2h*)(d + 1 * C) = (v2h){(_Float16)(p0 + r0),       (_Float16)(p1 + r1)};
            *(v2h*)(d + 2 * C) = (v2h){(_Float16)(q0 + s0),       (_Float16)(q1 + s1)};
            *(v2h*)(d + 3 * C) = (v2h){(_Float16)fabsf(p0 - r0),  (_Float16)fabsf(p1 - r1)};
            *(v2h*)(d + 4 * C) = (v2h){(_Float16)fabsf(q0 - s0),  (_Float16)fabsf(q1 - s1)};
        }
    }
#if USE_ASYNC_LDS
    __builtin_amdgcn_s_wait_asynccnt(0);               // W panel landed in LDS
#endif
    __syncthreads();

    if (valid) {
        v8f acc0 = {}, acc1 = {}, acc2 = {}, acc3 = {};
        const int row  = lane & 15;
        const int half = lane >> 4;
        const _Float16* arow = a_lds + (wave * 16 + row) * K;

        // A (16-bit 16x32): lanes 0-15 hold K 0..15; lanes 16-31 hold
        // K 16..31. B: lanes 0-15 col N, K 0..15; lanes 16-31 K 16..31.
        constexpr int KSTEPS = K / 32;
#pragma unroll
        for (int kk = 0; kk < KSTEPS; ++kk) {
            const int ka = kk * 32 + half * 16;
            V16HU a;
            a.u[0] = *(const uint4*)(arow + ka);
            a.u[1] = *(const uint4*)(arow + ka + 8);

            const int kb = kk * 32 + half * 16;
            const _Float16* wbase = w_lds + row * K + kb;
            V16HU b0, b1, b2, b3;
            b0.u[0] = *(const uint4*)(wbase + 0 * 16 * K);
            b0.u[1] = *(const uint4*)(wbase + 0 * 16 * K + 8);
            b1.u[0] = *(const uint4*)(wbase + 1 * 16 * K);
            b1.u[1] = *(const uint4*)(wbase + 1 * 16 * K + 8);
            b2.u[0] = *(const uint4*)(wbase + 2 * 16 * K);
            b2.u[1] = *(const uint4*)(wbase + 2 * 16 * K + 8);
            b3.u[0] = *(const uint4*)(wbase + 3 * 16 * K);
            b3.u[1] = *(const uint4*)(wbase + 3 * 16 * K + 8);

            acc0 = __builtin_amdgcn_wmma_f32_16x16x32_f16(false, a.v, false, b0.v,
                                                          (short)0, acc0, false, false);
            acc1 = __builtin_amdgcn_wmma_f32_16x16x32_f16(false, a.v, false, b1.v,
                                                          (short)0, acc1, false, false);
            acc2 = __builtin_amdgcn_wmma_f32_16x16x32_f16(false, a.v, false, b2.v,
                                                          (short)0, acc2, false, false);
            acc3 = __builtin_amdgcn_wmma_f32_16x16x32_f16(false, a.v, false, b3.v,
                                                          (short)0, acc3, false, false);
        }

        // D layout: lane (&15) -> N, VGPR r -> M = (lane>=16 ? 8 : 0) + r.
        const int moff = half * 8;
        float* ybm = y + (size_t)b * COUT * E_ + e0 + moff;
        v8f accs[4] = {acc0, acc1, acc2, acc3};
#pragma unroll
        for (int nt = 0; nt < 4; ++nt) {
            const int o = nt * 16 + row;
            const float bv = bias[o];
            float* yo = ybm + (size_t)o * E_;
            float4 lo4, hi4;
            lo4.x = accs[nt][0] + bv; lo4.y = accs[nt][1] + bv;
            lo4.z = accs[nt][2] + bv; lo4.w = accs[nt][3] + bv;
            hi4.x = accs[nt][4] + bv; hi4.y = accs[nt][5] + bv;
            hi4.z = accs[nt][6] + bv; hi4.w = accs[nt][7] + bv;
            *(float4*)(yo)     = lo4;
            *(float4*)(yo + 4) = hi4;
        }
    }
}

// ---------------------------------------------------------------------------
// Per-(b,o) mean / rstd over E.
// ---------------------------------------------------------------------------
__global__ void stats_kernel(const float* __restrict__ y, float2* __restrict__ st) {
    const int row = blockIdx.x;                 // b*64 + o
    const float* p = y + (size_t)row * E_;
    float s = 0.f, q = 0.f;
    for (int i = threadIdx.x; i < E_; i += blockDim.x) {
        const float v = p[i];
        s += v; q += v * v;
    }
    __shared__ float ls[256], lq[256];
    ls[threadIdx.x] = s; lq[threadIdx.x] = q;
    __syncthreads();
    for (int off = 128; off > 0; off >>= 1) {
        if (threadIdx.x < off) {
            ls[threadIdx.x] += ls[threadIdx.x + off];
            lq[threadIdx.x] += lq[threadIdx.x + off];
        }
        __syncthreads();
    }
    if (threadIdx.x == 0) {
        const float m = ls[0] / (float)E_;
        float v = lq[0] / (float)E_ - m * m;
        st[row] = make_float2(m, rsqrtf(v + EPS_));
    }
}

// ---------------------------------------------------------------------------
// out = relu((y - mean) * rstd [+ res])
// ---------------------------------------------------------------------------
__global__ void norm_relu_kernel(const float* __restrict__ y,
                                 const float2* __restrict__ st,
                                 const float* __restrict__ res,
                                 float* __restrict__ out) {
    const size_t n = (size_t)B_ * COUT * E_;
    const size_t i = (size_t)blockIdx.x * blockDim.x + threadIdx.x;
    if (i >= n) return;
    const int row = (int)(i / E_);
    const float2 mr = st[row];
    float v = (y[i] - mr.x) * mr.y;
    if (res) v += res[i];
    out[i] = fmaxf(v, 0.f);
}

// ---------------------------------------------------------------------------
extern "C" void kernel_launch(void* const* d_in, const int* in_sizes, int n_in,
                              void* d_out, int out_size, void* d_ws, size_t ws_size,
                              hipStream_t stream) {
    const float* fe      = (const float*)d_in[0];   // [4][32][E]
    const int*   eidx    = (const int*)  d_in[1];   // [4][E][4]
    const float* conv1_w = (const float*)d_in[2];   // [64][32][5]
    const float* conv1_b = (const float*)d_in[3];   // [64]
    const float* conv2_w = (const float*)d_in[4];   // [2][64][64][5]
    const float* conv2_b = (const float*)d_in[5];   // [2][64]
    float* out = (float*)d_out;                     // [4][64][E]

    char* ws = (char*)d_ws;
    const size_t K1 = 32 * 5, K2 = 64 * 5;
    _Float16* wp1  = (_Float16*)(ws);                               // 64*160 halves
    _Float16* wp2a = (_Float16*)(ws + 64 * K1 * 2);                 // 64*320 halves
    _Float16* wp2b = wp2a + 64 * K2;
    size_t off = 64 * K1 * 2 + 2 * 64 * K2 * 2;                     // 102400 B
    off = (off + 255) & ~(size_t)255;
    float*  ybuf = (float*)(ws + off);                              // [4][64][E]
    off += (size_t)B_ * COUT * E_ * 4;
    float*  x1   = (float*)(ws + off);                              // [4][64][E]
    off += (size_t)B_ * COUT * E_ * 4;
    float2* st   = (float2*)(ws + off);                             // 256 rows

    const dim3 convBlk(NW * 32);
    const dim3 convGrid((E_ / 16 + NW - 1) / NW, B_);
    const size_t shm1 = (64 * K1 + NW * 16 * K1) * 2 + NW * 16 * 4 * 4;
    const size_t shm2 = (64 * K2 + NW * 16 * K2) * 2 + NW * 16 * 4 * 4;
    const size_t nElem = (size_t)B_ * COUT * E_;
    const int ewGrid = (int)((nElem + 255) / 256);

    // pack weights
    pack_w_kernel<<<(64 * 32 * 5 + 255) / 256, 256, 0, stream>>>(conv1_w, wp1, 64, 32);
    pack_w_kernel<<<(64 * 64 * 5 + 255) / 256, 256, 0, stream>>>(conv2_w, wp2a, 64, 64);
    pack_w_kernel<<<(64 * 64 * 5 + 255) / 256, 256, 0, stream>>>(conv2_w + 64 * 64 * 5, wp2b, 64, 64);

    // conv1: fe (C=32) -> y ; norm+relu -> x1
    mesh_conv_wmma<32><<<convGrid, convBlk, shm1, stream>>>(fe, eidx, wp1, conv1_b, ybuf);
    stats_kernel<<<B_ * COUT, 256, 0, stream>>>(ybuf, st);
    norm_relu_kernel<<<ewGrid, 256, 0, stream>>>(ybuf, st, nullptr, x1);

    // block 0: x1 -> y ; relu(norm(y) + x1) -> d_out (x2)
    mesh_conv_wmma<64><<<convGrid, convBlk, shm2, stream>>>(x1, eidx, wp2a, conv2_b, ybuf);
    stats_kernel<<<B_ * COUT, 256, 0, stream>>>(ybuf, st);
    norm_relu_kernel<<<ewGrid, 256, 0, stream>>>(ybuf, st, x1, out);

    // block 1: d_out -> y ; relu(norm(y) + d_out) -> d_out
    mesh_conv_wmma<64><<<convGrid, convBlk, shm2, stream>>>(out, eidx, wp2b, conv2_b + 64, ybuf);
    stats_kernel<<<B_ * COUT, 256, 0, stream>>>(ybuf, st);
    norm_relu_kernel<<<ewGrid, 256, 0, stream>>>(ybuf, st, out, out);
}
